// TransformerBlock_74878459838741
// MI455X (gfx1250) — compile-verified, matrix operand order checked by reading, not run on hardware
//
#include <hip/hip_runtime.h>
#include <math.h>

// ---------------------------------------------------------------------------
// Types for CDNA5 WMMA (wave32): v_wmma_f32_16x16x32_bf16
// ---------------------------------------------------------------------------
typedef __bf16 bf16_t;
typedef bf16_t v16bf __attribute__((ext_vector_type(16)));
typedef bf16_t v8bf  __attribute__((ext_vector_type(8)));
typedef float  v8f   __attribute__((ext_vector_type(8)));

static constexpr int Bm = 2, Tm = 2048, Cm = 1024, Hm = 16, HDm = 64;
static constexpr int BT = Bm * Tm;           // 4096 rows
static constexpr int C3 = 3 * Cm;            // 3072

// fp32 -> bf16 round-to-nearest-even (bitwise, no reliance on cvt codegen)
__device__ __forceinline__ bf16_t f2bf(float f) {
    union { float f; unsigned u; } v; v.f = f;
    unsigned r = v.u + 0x7FFFu + ((v.u >> 16) & 1u);
    unsigned short h = (unsigned short)(r >> 16);
    return __builtin_bit_cast(bf16_t, h);
}

// Load a 16x32 bf16 fragment in the documented A-matrix layout:
//   lanes 0-15: rows 0-15, K = 0..7 (vgpr0-3) and 16..23 (vgpr4-7)
//   lanes 16-31: rows 0-15, K = 8..15 and 24..31
// `p` points at element [row0, k0] of a row-major matrix with leading dim ld.
// Also used for B fragments: out = A*W^T with W row-major [N][K] means the
// B operand loads with the identical pattern (lane = N row, K striped).
__device__ __forceinline__ v16bf load_frag(const bf16_t* __restrict__ p,
                                           int ld, int lane) {
    const int r    = lane & 15;
    const int koff = (lane >> 4) << 3;           // 0 or 8
    const bf16_t* q = p + (size_t)r * ld + koff;
    v8bf lo = *reinterpret_cast<const v8bf*>(q);
    v8bf hi = *reinterpret_cast<const v8bf*>(q + 16);
    v16bf f;
#pragma unroll
    for (int i = 0; i < 8; ++i) { f[i] = lo[i]; f[i + 8] = hi[i]; }
    return f;
}

__device__ __forceinline__ v8f wmma_bf16(v16bf a, v16bf b, v8f c) {
    return __builtin_amdgcn_wmma_f32_16x16x32_bf16(
        /*neg_a=*/false, a, /*neg_b=*/false, b,
        /*c_mod=*/(short)0, c, /*reuse_a=*/false, /*reuse_b=*/false);
}

// ---------------------------------------------------------------------------
// Tensor Data Mover staging (gfx1250). Guarded so the file compiles on any
// toolchain; falls back to coalesced v8bf global->LDS copies.
// ---------------------------------------------------------------------------
#if defined(__gfx1250__) && __has_builtin(__builtin_amdgcn_tensor_load_to_lds) \
    && __has_builtin(__builtin_amdgcn_s_wait_tensorcnt)
#define USE_TDM 1
#else
#define USE_TDM 0
#endif

#if USE_TDM
typedef unsigned u32x4 __attribute__((ext_vector_type(4)));
typedef int      i32x8 __attribute__((ext_vector_type(8)));
typedef int      i32x4 __attribute__((ext_vector_type(4)));

// 2-D tile load: `rows` x 32 bf16 elements, row stride = ldElems, into LDS
// at ldsDst (contiguous rows of 32 elems). D# built per ISA 08_async_tensor §8.
__device__ __forceinline__ void tdm_load_tile(bf16_t* ldsDst,
                                              const bf16_t* gsrc,
                                              int ldElems, int rows) {
    const unsigned long long ga = (unsigned long long)(size_t)gsrc;
    u32x4 g0;
    g0[0] = 1u;                                   // count=1, user descriptor
    g0[1] = (unsigned)(size_t)ldsDst;             // lds_addr (byte offset)
    g0[2] = (unsigned)(ga & 0xFFFFFFFFu);         // global_addr[31:0]
    g0[3] = (unsigned)((ga >> 32) & 0x01FFFFFFu)  // global_addr[56:32]
            | 0x80000000u;                        // type=2 ("image")
    i32x8 g1;
    g1[0] = 0x00010000;                           // data_size=1 (2 bytes)
    g1[1] = (int)(((unsigned)ldElems & 0xFFFFu) << 16);          // tensor_dim0 lo
    g1[2] = (int)((((unsigned)ldElems >> 16) & 0xFFFFu)          // tensor_dim0 hi
                  | (((unsigned)rows & 0xFFFFu) << 16));         // tensor_dim1 lo
    g1[3] = (int)(32u << 16);                     // tile_dim0 = 32
    g1[4] = rows;                                 // tile_dim1 = rows, tile_dim2=0
    g1[5] = ldElems;                              // tensor_dim0_stride lo32
    g1[6] = 0;                                    // stride hi / dim1_stride lo
    g1[7] = 0;                                    // dim1_stride hi
    i32x4 z4 = {};
#if __clang_major__ >= 23
    i32x8 z8 = {};
    __builtin_amdgcn_tensor_load_to_lds(g0, g1, z4, z4, z8, 0);
#else
    __builtin_amdgcn_tensor_load_to_lds(g0, g1, z4, z4, 0);
#endif
}
#endif

// ---------------------------------------------------------------------------
// fp32 -> bf16 conversion (weights)
// ---------------------------------------------------------------------------
__global__ void cvt_kernel(const float* __restrict__ in,
                           bf16_t* __restrict__ out, int n) {
    for (int i = blockIdx.x * blockDim.x + threadIdx.x; i < n;
         i += gridDim.x * blockDim.x)
        out[i] = f2bf(in[i]);
}

// ---------------------------------------------------------------------------
// LayerNorm: one 256-thread block per row of C=1024; fp32 in, bf16 out
// ---------------------------------------------------------------------------
__global__ __launch_bounds__(256)
void ln_kernel(const float* __restrict__ x, const float* __restrict__ w,
               const float* __restrict__ b, bf16_t* __restrict__ out) {
    const int row = blockIdx.x;
    const float* xr = x + (size_t)row * Cm;
    float s = 0.f, s2 = 0.f;
    for (int i = threadIdx.x; i < Cm; i += 256) {
        float v = xr[i]; s += v; s2 += v * v;
    }
#pragma unroll
    for (int o = 16; o; o >>= 1) {
        s  += __shfl_xor(s,  o, 32);
        s2 += __shfl_xor(s2, o, 32);
    }
    __shared__ float red[2][8];
    const int wv = threadIdx.x >> 5, ln = threadIdx.x & 31;
    if (ln == 0) { red[0][wv] = s; red[1][wv] = s2; }
    __syncthreads();
    s = 0.f; s2 = 0.f;
#pragma unroll
    for (int i = 0; i < 8; ++i) { s += red[0][i]; s2 += red[1][i]; }
    const float mu  = s * (1.0f / Cm);
    const float var = s2 * (1.0f / Cm) - mu * mu;
    const float rs  = rsqrtf(var + 1e-5f);
    for (int i = threadIdx.x; i < Cm; i += 256)
        out[(size_t)row * Cm + i] = f2bf((xr[i] - mu) * rs * w[i] + b[i]);
}

// ---------------------------------------------------------------------------
// Shared compute step for the GEMM: one 32-wide k-slab from LDS tiles.
// Wave owns a 64(M) x 32(N) output patch = 4x2 WMMA accumulators.
// ---------------------------------------------------------------------------
__device__ __forceinline__ void gemm_step(const bf16_t* __restrict__ Ab,
                                          const bf16_t* __restrict__ Bb,
                                          v8f (&acc)[4][2],
                                          int mw, int nw, int lane) {
    v16bf bF[2];
#pragma unroll
    for (int j = 0; j < 2; ++j)
        bF[j] = load_frag(Bb + (nw * 32 + j * 16) * 32, 32, lane);
#pragma unroll
    for (int i = 0; i < 4; ++i) {
        v16bf aF = load_frag(Ab + (mw * 64 + i * 16) * 32, 32, lane);
#pragma unroll
        for (int j = 0; j < 2; ++j)
            acc[i][j] = wmma_bf16(aF, bF[j], acc[i][j]);
    }
}

// ---------------------------------------------------------------------------
// WMMA GEMM: out[M,N] = A[M,K] * W[N,K]^T + bias, epilogue variants.
//   EPI 0: bf16 out    EPI 1: bf16 out + exact GELU    EPI 2: fp32 + residual
// Block = 256 threads = 8 waves, block tile 128(M) x 128(N), K-step 32.
// Double-buffered LDS software pipeline: the Tensor Data Mover fills tile t+1
// (async, TENSORcnt) while all 8 waves run WMMAs on tile t; a single
// s_wait_tensorcnt + barrier per k-step publishes the next buffer.
// Fallback (no TDM builtin): identical pipeline with register-held global
// loads issued before compute and LDS stores after.
// ---------------------------------------------------------------------------
template <int EPI>
__global__ __launch_bounds__(256)
void gemm_kernel(const bf16_t* __restrict__ A, const bf16_t* __restrict__ W,
                 const float* __restrict__ bias, const float* __restrict__ res,
                 float* __restrict__ outF, bf16_t* __restrict__ outB,
                 int M, int N, int K) {
    __shared__ __align__(16) bf16_t Abuf[2][128 * 32];
    __shared__ __align__(16) bf16_t Bbuf[2][128 * 32];
    const int lane = threadIdx.x & 31;
    const int wave = threadIdx.x >> 5;
    const int mw = wave >> 2;                 // 0..1
    const int nw = wave & 3;                  // 0..3
    const int m0 = blockIdx.y * 128;
    const int n0 = blockIdx.x * 128;
    const int nk = K >> 5;                    // k-steps of 32
    v8f acc[4][2] = {};

#if USE_TDM
    if (wave == 0) {
        tdm_load_tile(Abuf[0], A + (size_t)m0 * K, K, 128);
        tdm_load_tile(Bbuf[0], W + (size_t)n0 * K, K, 128);
        __builtin_amdgcn_s_wait_tensorcnt(0);
    }
    __syncthreads();
    int cur = 0;
    for (int t = 0; t < nk; ++t) {
        const int nxt = cur ^ 1;
        if (wave == 0 && t + 1 < nk) {        // async fill of next tile
            tdm_load_tile(Abuf[nxt], A + (size_t)m0 * K + (t + 1) * 32, K, 128);
            tdm_load_tile(Bbuf[nxt], W + (size_t)n0 * K + (t + 1) * 32, K, 128);
        }
        gemm_step(Abuf[cur], Bbuf[cur], acc, mw, nw, lane);  // overlapped
        if (wave == 0) __builtin_amdgcn_s_wait_tensorcnt(0);
        __syncthreads();
        cur = nxt;
    }
#else
    const int r = threadIdx.x >> 1;              // 0..127
    const int c = (threadIdx.x & 1) * 16;        // 0 or 16
    const bf16_t* ap = A + (size_t)(m0 + r) * K + c;
    const bf16_t* wp = W + (size_t)(n0 + r) * K + c;
    {
        v8bf a0 = *reinterpret_cast<const v8bf*>(ap);
        v8bf a1 = *reinterpret_cast<const v8bf*>(ap + 8);
        v8bf b0 = *reinterpret_cast<const v8bf*>(wp);
        v8bf b1 = *reinterpret_cast<const v8bf*>(wp + 8);
        *reinterpret_cast<v8bf*>(&Abuf[0][r * 32 + c])     = a0;
        *reinterpret_cast<v8bf*>(&Abuf[0][r * 32 + c + 8]) = a1;
        *reinterpret_cast<v8bf*>(&Bbuf[0][r * 32 + c])     = b0;
        *reinterpret_cast<v8bf*>(&Bbuf[0][r * 32 + c + 8]) = b1;
    }
    __syncthreads();
    int cur = 0;
    for (int t = 0; t < nk; ++t) {
        const int nxt = cur ^ 1;
        v8bf a0, a1, b0, b1;
        if (t + 1 < nk) {                        // loads in flight over compute
            const bf16_t* an = ap + (t + 1) * 32;
            const bf16_t* wn = wp + (t + 1) * 32;
            a0 = *reinterpret_cast<const v8bf*>(an);
            a1 = *reinterpret_cast<const v8bf*>(an + 8);
            b0 = *reinterpret_cast<const v8bf*>(wn);
            b1 = *reinterpret_cast<const v8bf*>(wn + 8);
        }
        gemm_step(Abuf[cur], Bbuf[cur], acc, mw, nw, lane);
        if (t + 1 < nk) {
            *reinterpret_cast<v8bf*>(&Abuf[nxt][r * 32 + c])     = a0;
            *reinterpret_cast<v8bf*>(&Abuf[nxt][r * 32 + c + 8]) = a1;
            *reinterpret_cast<v8bf*>(&Bbuf[nxt][r * 32 + c])     = b0;
            *reinterpret_cast<v8bf*>(&Bbuf[nxt][r * 32 + c + 8]) = b1;
        }
        __syncthreads();
        cur = nxt;
    }
#endif

    const int hi = lane >> 4;
#pragma unroll
    for (int i = 0; i < 4; ++i) {
#pragma unroll
        for (int j = 0; j < 2; ++j) {
            const int col = n0 + nw * 32 + j * 16 + (lane & 15);
            const float bv = bias[col];
#pragma unroll
            for (int e = 0; e < 8; ++e) {
                const int row = m0 + mw * 64 + i * 16 + e + hi * 8;
                float v = acc[i][j][e] + bv;
                if (EPI == 1) v = 0.5f * v * (1.0f + erff(v * 0.70710678118f));
                const size_t idx = (size_t)row * N + col;
                if (EPI == 2) outF[idx] = res[idx] + v;
                else          outB[idx] = f2bf(v);
            }
        }
    }
}

// ---------------------------------------------------------------------------
// Causal flash attention, WMMA tiles, online softmax.
// Grid: (T/128, B*H). Block = 256 threads = 8 waves; wave -> 16 query rows.
// qkv is bf16 [BT, 3C]; q at col h*64, k at C + h*64, v at 2C + h*64.
// ---------------------------------------------------------------------------
__global__ __launch_bounds__(256)
void attn_kernel(const bf16_t* __restrict__ qkv, bf16_t* __restrict__ y) {
    __shared__ __align__(16) bf16_t Kt[32 * 64];      // [key][hd]
    __shared__ __align__(16) bf16_t Vt[64 * 32];      // [hd][key] (transposed)
    __shared__ __align__(16) bf16_t Pl[8][16 * 32];   // per-wave P tile

    const int tid  = threadIdx.x;
    const int lane = tid & 31;
    const int wave = tid >> 5;
    const int hi   = lane >> 4;
    const int bh = blockIdx.y;
    const int b  = bh >> 4;      // H = 16
    const int h  = bh & 15;
    const int q0b = blockIdx.x * 128;
    const int q0w = q0b + wave * 16;

    const size_t qrow = ((size_t)(b * Tm + q0w)) * C3 + h * HDm;
    const v16bf qA0 = load_frag(qkv + qrow,       C3, lane);
    const v16bf qA1 = load_frag(qkv + qrow + 32,  C3, lane);

    v8f acc[4] = {};
    float mrun[8], lrun[8];
#pragma unroll
    for (int e = 0; e < 8; ++e) { mrun[e] = -1e30f; lrun[e] = 0.f; }

    const size_t kvbase = ((size_t)(b * Tm)) * C3 + h * HDm;

    for (int kb = 0; kb < q0b + 128; kb += 32) {
        __syncthreads();
        {   // cooperative staging: 32x64 K tile + transposed 64x32 V tile
            const int r = tid >> 3;             // key row 0..31
            const int c = (tid & 7) * 8;        // hd col 0..56
            const size_t base = kvbase + (size_t)(kb + r) * C3;
            v8bf kc = *reinterpret_cast<const v8bf*>(qkv + base + Cm + c);
            *reinterpret_cast<v8bf*>(&Kt[r * 64 + c]) = kc;
            v8bf vc = *reinterpret_cast<const v8bf*>(qkv + base + 2 * Cm + c);
#pragma unroll
            for (int i = 0; i < 8; ++i) Vt[(c + i) * 32 + r] = vc[i];
        }
        __syncthreads();

        // S = Q K^T  (two 16x16 tiles over the 32 keys)
        v8f s[2];
#pragma unroll
        for (int j = 0; j < 2; ++j) {
            v8f t = {};
            v16bf k0 = load_frag(Kt + j * 16 * 64,      64, lane);
            v16bf k1 = load_frag(Kt + j * 16 * 64 + 32, 64, lane);
            t = wmma_bf16(qA0, k0, t);
            t = wmma_bf16(qA1, k1, t);
            s[j] = t;
        }

        // scale, causal mask, row-max (reduce across 16-lane halves)
        float rmax[8];
#pragma unroll
        for (int e = 0; e < 8; ++e) {
            const int row = q0w + e + hi * 8;
#pragma unroll
            for (int j = 0; j < 2; ++j) {
                const int key = kb + j * 16 + (lane & 15);
                float v = s[j][e] * 0.125f;   // 1/sqrt(64)
                if (key > row) v = -1e30f;
                s[j][e] = v;
            }
            float rm = fmaxf(s[0][e], s[1][e]);
#pragma unroll
            for (int o = 8; o; o >>= 1) rm = fmaxf(rm, __shfl_xor(rm, o, 32));
            rmax[e] = rm;
        }

        // online softmax update; stash P (bf16) in per-wave LDS
#pragma unroll
        for (int e = 0; e < 8; ++e) {
            const float mn = fmaxf(mrun[e], rmax[e]);
            const float al = __expf(mrun[e] - mn);
            const float p0 = __expf(s[0][e] - mn);
            const float p1 = __expf(s[1][e] - mn);
            float rs = p0 + p1;
#pragma unroll
            for (int o = 8; o; o >>= 1) rs += __shfl_xor(rs, o, 32);
            lrun[e] = lrun[e] * al + rs;
            mrun[e] = mn;
            const int pr = (e + hi * 8) * 32 + (lane & 15);
            Pl[wave][pr]      = f2bf(p0);
            Pl[wave][pr + 16] = f2bf(p1);
#pragma unroll
            for (int j2 = 0; j2 < 4; ++j2) acc[j2][e] *= al;
        }

        // acc += P * V   (A = P 16x32 from LDS, B = V^T tiles 32x16)
        v16bf pA = load_frag(&Pl[wave][0], 32, lane);
#pragma unroll
        for (int j2 = 0; j2 < 4; ++j2) {
            v16bf vF = load_frag(Vt + j2 * 16 * 32, 32, lane);
            acc[j2] = wmma_bf16(pA, vF, acc[j2]);
        }
    }

    // normalize and write y [BT, C] bf16
#pragma unroll
    for (int j2 = 0; j2 < 4; ++j2) {
#pragma unroll
        for (int e = 0; e < 8; ++e) {
            const int row = q0w + e + hi * 8;
            const float v = acc[j2][e] / lrun[e];
            y[((size_t)(b * Tm + row)) * Cm + h * HDm + j2 * 16 + (lane & 15)] =
                f2bf(v);
        }
    }
}

// ---------------------------------------------------------------------------
// Host-side orchestration
// ---------------------------------------------------------------------------
extern "C" void kernel_launch(void* const* d_in, const int* in_sizes, int n_in,
                              void* d_out, int out_size, void* d_ws, size_t ws_size,
                              hipStream_t stream) {
    const float* x     = (const float*)d_in[0];
    const float* ln1w  = (const float*)d_in[1];
    const float* ln1b  = (const float*)d_in[2];
    const float* wqkv  = (const float*)d_in[3];
    const float* bqkv  = (const float*)d_in[4];
    const float* wo    = (const float*)d_in[5];
    const float* bo    = (const float*)d_in[6];
    const float* ln2w  = (const float*)d_in[7];
    const float* ln2b  = (const float*)d_in[8];
    const float* wfc   = (const float*)d_in[9];
    const float* bfc   = (const float*)d_in[10];
    const float* wproj = (const float*)d_in[11];
    const float* bproj = (const float*)d_in[12];
    float* out = (float*)d_out;

    // workspace layout (256B aligned)
    char* base = (char*)d_ws;
    size_t off = 0;
    auto alloc = [&](size_t bytes) {
        char* p = base + off;
        off += (bytes + 255) & ~(size_t)255;
        return p;
    };
    bf16_t* wqkv_bf  = (bf16_t*)alloc((size_t)3 * Cm * Cm * 2);
    bf16_t* wo_bf    = (bf16_t*)alloc((size_t)Cm * Cm * 2);
    bf16_t* wfc_bf   = (bf16_t*)alloc((size_t)4 * Cm * Cm * 2);
    bf16_t* wproj_bf = (bf16_t*)alloc((size_t)4 * Cm * Cm * 2);
    bf16_t* h_bf     = (bf16_t*)alloc((size_t)BT * Cm * 2);
    bf16_t* qkv_bf   = (bf16_t*)alloc((size_t)BT * C3 * 2);
    bf16_t* y_bf     = (bf16_t*)alloc((size_t)BT * Cm * 2);
    float*  x1       = (float*) alloc((size_t)BT * Cm * 4);
    bf16_t* h2_bf    = (bf16_t*)alloc((size_t)BT * Cm * 2);
    bf16_t* fc_bf    = (bf16_t*)alloc((size_t)BT * 4 * Cm * 2);

    // 1) weights -> bf16
    cvt_kernel<<<2048, 256, 0, stream>>>(wqkv,  wqkv_bf,  3 * Cm * Cm);
    cvt_kernel<<<2048, 256, 0, stream>>>(wo,    wo_bf,    Cm * Cm);
    cvt_kernel<<<2048, 256, 0, stream>>>(wfc,   wfc_bf,   4 * Cm * Cm);
    cvt_kernel<<<2048, 256, 0, stream>>>(wproj, wproj_bf, 4 * Cm * Cm);

    // 2) LN1
    ln_kernel<<<BT, 256, 0, stream>>>(x, ln1w, ln1b, h_bf);

    // 3) QKV projection: [4096,1024] x [3072,1024]^T -> bf16 [4096,3072]
    gemm_kernel<0><<<dim3(C3 / 128, BT / 128), 256, 0, stream>>>(
        h_bf, wqkv_bf, bqkv, nullptr, nullptr, qkv_bf, BT, C3, Cm);

    // 4) causal flash attention -> y_bf [4096,1024]
    attn_kernel<<<dim3(Tm / 128, Bm * Hm), 256, 0, stream>>>(qkv_bf, y_bf);

    // 5) output projection + residual: x1 = x + y @ wo^T + bo  (fp32)
    gemm_kernel<2><<<dim3(Cm / 128, BT / 128), 256, 0, stream>>>(
        y_bf, wo_bf, bo, x, x1, nullptr, BT, Cm, Cm);

    // 6) LN2
    ln_kernel<<<BT, 256, 0, stream>>>(x1, ln2w, ln2b, h2_bf);

    // 7) FC + exact GELU -> bf16 [4096,4096]
    gemm_kernel<1><<<dim3(4 * Cm / 128, BT / 128), 256, 0, stream>>>(
        h2_bf, wfc_bf, bfc, nullptr, nullptr, fc_bf, BT, 4 * Cm, Cm);

    // 8) down projection + residual -> fp32 out
    gemm_kernel<2><<<dim3(Cm / 128, BT / 128), 256, 0, stream>>>(
        fc_bf, wproj_bf, bproj, x1, out, nullptr, BT, Cm, 4 * Cm);

    (void)in_sizes; (void)n_in; (void)out_size; (void)ws_size;
}
